// LSTMBijection_10273561772170
// MI455X (gfx1250) — matrix operand order repaired and checked by reading, not compiled, not run on hardware
//
#include <hip/hip_runtime.h>
#include <hip/hip_bf16.h>
#include <math.h>

#define VV 32000
#define DD 192
#define LL 6
#define BB 32
#define TT 128
#define HH4 768
#define MM (BB * TT) /* 4096 tokens */

#define MBLK 128
#define NBLK 64

typedef __attribute__((ext_vector_type(16))) _Float16 v16h;
typedef __attribute__((ext_vector_type(8)))  _Float16 v8h;
typedef __attribute__((ext_vector_type(8)))  float    v8f;

#ifndef __has_builtin
#define __has_builtin(x) 0
#endif

#if __has_builtin(__builtin_amdgcn_global_load_async_to_lds_b128)
#define HAVE_ASYNC_LDS 1
#else
#define HAVE_ASYNC_LDS 0
#endif

// Builtin parameter types per hipcc diagnostic: pointer-to-int4 (vector_size 16)
typedef int v4i_ __attribute__((vector_size(16)));
typedef __attribute__((address_space(1))) v4i_ as1_v4i;
typedef __attribute__((address_space(3))) v4i_ as3_v4i;

// 16-byte global -> LDS copy; async (ASYNCcnt-tracked) on gfx1250.
__device__ __forceinline__ void copy_b128_to_lds(const _Float16* g, _Float16* l) {
#if HAVE_ASYNC_LDS
    __builtin_amdgcn_global_load_async_to_lds_b128((as1_v4i*)g, (as3_v4i*)l, 0, 0);
#else
    *(v8h*)l = *(const v8h*)g;
#endif
}

__device__ __forceinline__ void copy_wait() {
#if HAVE_ASYNC_LDS
#if __has_builtin(__builtin_amdgcn_s_wait_asynccnt)
    __builtin_amdgcn_s_wait_asynccnt(0);
#else
    asm volatile("s_wait_asynccnt 0x0" ::: "memory");
#endif
#endif
}

// ---------------------------------------------------------------------------
// WMMA fragment loaders (CDNA5 16x16x32 f16, wave32; ISA 7.12.2 layouts)
// A: 16x32 (MxK). lane<16 holds M=lane, K={0..7,16..23}; lane>=16 K={8..15,24..31}
// ---------------------------------------------------------------------------
__device__ __forceinline__ v16h load_a_frag(const _Float16* A, int lda, int lane) {
    int row = lane & 15;
    int kh  = (lane >> 4) << 3;              // 0 or 8
    const _Float16* p = A + row * lda + kh;
    v8h lo = *(const v8h*)(p);               // K = kh .. kh+7
    v8h hi = *(const v8h*)(p + 16);          // K = kh+16 .. kh+23
    return __builtin_shufflevector(lo, hi, 0, 1, 2, 3, 4, 5, 6, 7,
                                           8, 9, 10, 11, 12, 13, 14, 15);
}

// B: 32x16 (KxN) from row-major weight [N, K] (i.e. B^T in memory).
// lane<16: n=lane, K={0..7,16..23} ; lane>=16: n=lane-16, K={8..15,24..31}
__device__ __forceinline__ v16h load_b_frag(const _Float16* Bt, int ldb, int lane) {
    int col = lane & 15;
    int kh  = (lane >> 4) << 3;              // 0 or 8
    const _Float16* p = Bt + col * ldb + kh;
    v8h lo = *(const v8h*)(p);
    v8h hi = *(const v8h*)(p + 16);
    return __builtin_shufflevector(lo, hi, 0, 1, 2, 3, 4, 5, 6, 7,
                                           8, 9, 10, 11, 12, 13, 14, 15);
}

// ---------------------------------------------------------------------------
// Small helper kernels
// ---------------------------------------------------------------------------
__global__ void k_f32_to_f16(const float* __restrict__ s, _Float16* __restrict__ d, int n) {
    int i = blockIdx.x * blockDim.x + threadIdx.x;
    if (i < n) d[i] = (_Float16)s[i];
}

__global__ void k_bias_sum(const float* __restrict__ a, const float* __restrict__ b,
                           float* __restrict__ d, int n) {
    int i = blockIdx.x * blockDim.x + threadIdx.x;
    if (i < n) d[i] = a[i] + b[i];
}

__global__ void k_embed(const int* __restrict__ x, const float* __restrict__ te,
                        const float* __restrict__ pe, float* __restrict__ h32,
                        _Float16* __restrict__ h16) {
    int i = blockIdx.x * blockDim.x + threadIdx.x;
    if (i >= MM * DD) return;
    int m = i / DD, d = i - m * DD;
    int t = m & (TT - 1);
    int tok = x[m];
    float v = te[(size_t)tok * DD + d] + pe[(size_t)t * DD + d];
    h32[i] = v;
    h16[i] = (_Float16)v;
}

// ---------------------------------------------------------------------------
// Tiled GEMM: C[M,N] = A[M,K=192] * B[N,K]^T (+ bias[n]); f16 in, f32 out.
// Block = 256 thr (8 waves) computes a 128x64 tile. Both operand tiles staged
// in LDS via async global->LDS copies; wave w owns M-tile w (4 N-tiles each).
// ---------------------------------------------------------------------------
__global__ void __launch_bounds__(256) k_gemm_tiled(const _Float16* __restrict__ A,
                                                    const _Float16* __restrict__ Bt,
                                                    const float* __restrict__ bias,
                                                    float* __restrict__ C, int N) {
    extern __shared__ __attribute__((aligned(32))) _Float16 smem[];
    _Float16* As = smem;                 // [128][192] f16 = 48 KB
    _Float16* Bs = smem + MBLK * DD;     // [64][192]  f16 = 24 KB
    int tid = threadIdx.x;
    int lane = tid & 31, w = tid >> 5;
    int m0 = blockIdx.x * MBLK;
    int n0 = blockIdx.y * NBLK;

    const _Float16* Ag = A + (size_t)m0 * DD;
    for (int i = tid; i < MBLK * DD / 8; i += 256)
        copy_b128_to_lds(Ag + (size_t)i * 8, As + (size_t)i * 8);
    const _Float16* Bg = Bt + (size_t)n0 * DD;
    for (int i = tid; i < NBLK * DD / 8; i += 256)
        copy_b128_to_lds(Bg + (size_t)i * 8, Bs + (size_t)i * 8);
    copy_wait();          // drain this wave's ASYNCcnt
    __syncthreads();      // all waves' copies now visible

    const _Float16* Aw = As + (size_t)(w * 16) * DD;
    v8f acc[4] = {};
#pragma unroll
    for (int k0 = 0; k0 < DD; k0 += 32) {
        v16h a = load_a_frag(Aw + k0, DD, lane);
#pragma unroll
        for (int j = 0; j < 4; ++j) {
            v16h b = load_b_frag(Bs + (size_t)(j * 16) * DD + k0, DD, lane);
            acc[j] = __builtin_amdgcn_wmma_f32_16x16x32_f16(false, a, false, b,
                                                            (short)0, acc[j], false, false);
        }
    }

    int col = lane & 15, rb = (lane >> 4) << 3;
    int mrow = m0 + w * 16;
#pragma unroll
    for (int j = 0; j < 4; ++j) {
        int nn = n0 + (j << 4) + col;
        float bv = bias ? bias[nn] : 0.f;
#pragma unroll
        for (int r = 0; r < 8; ++r) {
            C[(size_t)(mrow + rb + r) * (size_t)N + nn] = acc[j][r] + bv;
        }
    }
}

// ---------------------------------------------------------------------------
// LSTM scan: 2 blocks x 512 threads; block handles 16 batch rows.
// Per timestep: G = h[16x192] * Whh^T[192x768] via WMMA (48 N-tiles over 16
// waves), gate nonlinearity + c/h update elementwise from LDS.
// ---------------------------------------------------------------------------
__global__ void __launch_bounds__(512) k_lstm_scan(const _Float16* __restrict__ Whh,
                                                   const float* __restrict__ gx,
                                                   float* __restrict__ h32,
                                                   _Float16* __restrict__ h16,
                                                   float* __restrict__ cbuf) {
    __shared__ __align__(32) _Float16 hA[16 * DD];      // 6 KB  (A fragments)
    __shared__ float gates[16 * HH4];                   // 48 KB (gate preacts)
    int tid = threadIdx.x;
    int lane = tid & 31, w = tid >> 5;
    int bb0 = blockIdx.x << 4;
    float* cb = cbuf + (size_t)bb0 * DD;

    for (int i = tid; i < 16 * DD; i += 512) { hA[i] = (_Float16)0.f; cb[i] = 0.f; }
    __syncthreads();

    for (int t = 0; t < TT; ++t) {
        // ---- recurrent matmul: each wave owns 3 N-tiles ----
#pragma unroll
        for (int j = 0; j < 3; ++j) {
            int n0 = (w * 3 + j) << 4;
            const _Float16* Bt = Whh + (size_t)n0 * DD;
            v8f acc = {};
#pragma unroll
            for (int k0 = 0; k0 < DD; k0 += 32) {
                __builtin_prefetch((const void*)(Bt + k0 + 32), 0, 0);
                v16h a = load_a_frag(hA + k0, DD, lane);     // ds_load from LDS
                v16h b = load_b_frag(Bt + k0, DD, lane);
                acc = __builtin_amdgcn_wmma_f32_16x16x32_f16(false, a, false, b,
                                                             (short)0, acc, false, false);
            }
            int col = lane & 15, rb = (lane >> 4) << 3;
#pragma unroll
            for (int r = 0; r < 8; ++r) {
                int row = rb + r;
                size_t m = (size_t)(bb0 + row) * TT + t;
                gates[row * HH4 + n0 + col] = acc[r] + gx[m * HH4 + n0 + col];
            }
        }
        __syncthreads();

        // ---- gate nonlinearity + state update (3072 elems / 512 thr) ----
        for (int i = tid; i < 16 * DD; i += 512) {
            int r = i / DD, d = i - r * DD;
            float gi = gates[r * HH4 + d];
            float gf = gates[r * HH4 + DD + d];
            float gg = gates[r * HH4 + 2 * DD + d];
            float go = gates[r * HH4 + 3 * DD + d];
            float si = 1.f / (1.f + __expf(-gi));
            float sf = 1.f / (1.f + __expf(-gf));
            float tg = tanhf(gg);
            float so = 1.f / (1.f + __expf(-go));
            float c = sf * cb[i] + si * tg;
            cb[i] = c;
            float h = so * tanhf(c);
            hA[r * DD + d] = (_Float16)h;
            size_t m = (size_t)(bb0 + r) * TT + t;
            h32[m * DD + d] = h;
            h16[m * DD + d] = (_Float16)h;
        }
        __syncthreads();
    }
}

// ---------------------------------------------------------------------------
// LayerNorm over D=192; writes f32 to d_out tail and f16 for the head GEMM.
// ---------------------------------------------------------------------------
__global__ void __launch_bounds__(256) k_layernorm(const float* __restrict__ h32,
                                                   const float* __restrict__ gamma,
                                                   const float* __restrict__ beta,
                                                   float* __restrict__ outh,
                                                   _Float16* __restrict__ h16) {
    int m = blockIdx.x;
    int tid = threadIdx.x;
    __shared__ float ssum[256];
    __shared__ float ssq[256];
    float v = 0.f;
    if (tid < DD) v = h32[(size_t)m * DD + tid];
    ssum[tid] = v;
    ssq[tid]  = v * v;
    __syncthreads();
    for (int s = 128; s > 0; s >>= 1) {
        if (tid < s) { ssum[tid] += ssum[tid + s]; ssq[tid] += ssq[tid + s]; }
        __syncthreads();
    }
    float mu  = ssum[0] / (float)DD;
    float var = ssq[0] / (float)DD - mu * mu;
    float inv = rsqrtf(var + 1e-5f);
    if (tid < DD) {
        float y = (v - mu) * inv * gamma[tid] + beta[tid];
        outh[(size_t)m * DD + tid] = y;
        h16[(size_t)m * DD + tid]  = (_Float16)y;
    }
}

// ---------------------------------------------------------------------------
// Host orchestration
// ---------------------------------------------------------------------------
extern "C" void kernel_launch(void* const* d_in, const int* in_sizes, int n_in,
                              void* d_out, int out_size, void* d_ws, size_t ws_size,
                              hipStream_t stream) {
    (void)in_sizes; (void)n_in; (void)out_size; (void)ws_size;
    const int*   x     = (const int*)d_in[0];
    const float* temb  = (const float*)d_in[1];
    const float* pemb  = (const float*)d_in[2];
    const float* Wih   = (const float*)d_in[3];
    const float* Whh   = (const float*)d_in[4];
    const float* bih   = (const float*)d_in[5];
    const float* bhh   = (const float*)d_in[6];
    const float* gamma = (const float*)d_in[7];
    const float* beta  = (const float*)d_in[8];

    float* out   = (float*)d_out;
    float* out_h = out + (size_t)MM * VV;

    char* ws = (char*)d_ws;
    size_t o = 0;
    auto take = [&](size_t nbytes) -> void* {
        void* p = ws + o;
        o += (nbytes + 255) & ~(size_t)255;
        return p;
    };
    float*     h32    = (float*)take((size_t)MM * DD * 4);
    _Float16*  h16    = (_Float16*)take((size_t)MM * DD * 2);
    float*     gx     = (float*)take((size_t)MM * HH4 * 4);
    _Float16*  Wih16  = (_Float16*)take((size_t)LL * HH4 * DD * 2);
    _Float16*  Whh16  = (_Float16*)take((size_t)LL * HH4 * DD * 2);
    float*     biasb  = (float*)take((size_t)LL * HH4 * 4);
    _Float16*  temb16 = (_Float16*)take((size_t)VV * DD * 2);
    float*     cbuf   = (float*)take((size_t)BB * DD * 4);

    const size_t smem_bytes = (size_t)(MBLK + NBLK) * DD * sizeof(_Float16); // 73728

    int n;
    n = VV * DD;
    k_f32_to_f16<<<(n + 255) / 256, 256, 0, stream>>>(temb, temb16, n);
    n = LL * HH4 * DD;
    k_f32_to_f16<<<(n + 255) / 256, 256, 0, stream>>>(Wih, Wih16, n);
    k_f32_to_f16<<<(n + 255) / 256, 256, 0, stream>>>(Whh, Whh16, n);
    n = LL * HH4;
    k_bias_sum<<<(n + 255) / 256, 256, 0, stream>>>(bih, bhh, biasb, n);
    n = MM * DD;
    k_embed<<<(n + 255) / 256, 256, 0, stream>>>(x, temb, pemb, h32, h16);

    for (int l = 0; l < LL; ++l) {
        k_gemm_tiled<<<dim3(MM / MBLK, HH4 / NBLK), 256, smem_bytes, stream>>>(
            h16, Wih16 + (size_t)l * HH4 * DD, biasb + (size_t)l * HH4, gx, HH4);
        k_lstm_scan<<<2, 512, 0, stream>>>(Whh16 + (size_t)l * HH4 * DD, gx,
                                           h32, h16, cbuf);
    }

    k_layernorm<<<MM, 256, 0, stream>>>(h32, gamma, beta, out_h, h16);

    k_gemm_tiled<<<dim3(MM / MBLK, VV / NBLK), 256, smem_bytes, stream>>>(
        h16, temb16, nullptr, out, VV);
}
